// MultiHeadAttention_49890340110619
// MI455X (gfx1250) — matrix operand order repaired and checked by reading, not compiled, not run on hardware
//
#include <hip/hip_runtime.h>
#include <hip/hip_bf16.h>

// Sliding-window MHA for MI455X (gfx1250), wave32 + v_wmma_f32_16x16x32_f16.
// Pipeline: [f32->f16 convert] -> [QKV gemm] -> [banded attention] -> [out-proj].
// Inner loops are software-pipelined with register double buffers; a
// sched_barrier between the next-chunk load group and the current-chunk WMMA
// group stops the scheduler from collapsing the pipeline to just-in-time loads.

typedef __attribute__((ext_vector_type(16))) _Float16 v16h;
typedef __attribute__((ext_vector_type(8)))  _Float16 v8h;
typedef __attribute__((ext_vector_type(8)))  float    v8f;

#define PIPELINE_FENCE() __builtin_amdgcn_sched_barrier(0)

// ---- problem constants (from reference setup_inputs) ----
constexpr int Bb   = 2;
constexpr int Ss   = 2048;
constexpr int Dd   = 768;    // == E
constexpr int Hh   = 12;
constexpr int HD   = 64;     // head dim
constexpr int N3E  = 2304;   // 3*E
constexpr int WIN  = 128;    // WINDOW_SIZE/2
constexpr int KT   = 17;     // key tiles per query tile: 16 + 2*128 = 272 keys
constexpr int KEYS = 272;
constexpr int KCH  = Dd / 32;  // 24 K-chunks in the projections

static __device__ inline v8f wmma16x16x32(v16h a, v16h b, v8f c) {
  return __builtin_amdgcn_wmma_f32_16x16x32_f16(false, a, false, b, (short)0, c,
                                                false, false);
}

static __device__ inline v8f zero8f() {
  v8f z;
#pragma unroll
  for (int i = 0; i < 8; ++i) z[i] = 0.0f;
  return z;
}
static __device__ inline v16h zero16h() {
  v16h z;
#pragma unroll
  for (int i = 0; i < 16; ++i) z[i] = (_Float16)0.0f;
  return z;
}
static __device__ inline v8h ld8(const _Float16* p) {  // 16B-aligned v8h load
  return *(const v8h*)p;
}
static __device__ inline v16h cat(v8h lo, v8h hi) {
  return __builtin_shufflevector(lo, hi, 0, 1, 2, 3, 4, 5, 6, 7,
                                 8, 9, 10, 11, 12, 13, 14, 15);
}

// A-operand (16-bit, 16x32 MxK). Lane l holds row m=l&15.
// Element e -> K = 16*(l>>4) + e: one contiguous run of 16 (two aligned v8h).
static __device__ inline v16h a_op(const _Float16* __restrict__ row, int hiK) {
  return cat(ld8(row + hiK * 16), ld8(row + hiK * 16 + 8));
}
// B-operand (16-bit, 32x16 KxN). Lane l holds col n=l&15, element e is
// K = 16*(l>>4) + e: one contiguous run of 16 halves (two aligned v8h).
static __device__ inline v16h b_op(const _Float16* __restrict__ p) {
  return cat(ld8(p), ld8(p + 8));
}

// ============================================================================
// Kernel 0: elementwise f32 -> f16 (vectorized, one-time cost ~12 MB traffic)
// ============================================================================
__global__ __launch_bounds__(256) void cvt_f16_kernel(
    const float* __restrict__ src, _Float16* __restrict__ dst, int n) {
  const int i = (blockIdx.x * 256 + threadIdx.x) * 8;
  if (i + 8 <= n) {
    const float4 a = *(const float4*)(src + i);
    const float4 b = *(const float4*)(src + i + 4);
    v8h h;
    h[0] = (_Float16)a.x; h[1] = (_Float16)a.y;
    h[2] = (_Float16)a.z; h[3] = (_Float16)a.w;
    h[4] = (_Float16)b.x; h[5] = (_Float16)b.y;
    h[6] = (_Float16)b.z; h[7] = (_Float16)b.w;
    *(v8h*)(dst + i) = h;
  }
}

// ============================================================================
// Kernel 1: qkv = x @ Wqkv^T + bqkv, scattered into Q[b,h,s,d], K[b,h,s,d],
// Vt[b,h,d,s] (V transposed so attention's AV B-operand loads are contiguous).
// One wave per 32(M) x 64(N) tile; 2-stage software pipeline over K=32 chunks.
// ============================================================================
__global__ __launch_bounds__(32) void qkv_gemm_kernel(
    const _Float16* __restrict__ xh, const _Float16* __restrict__ Wh,
    const float* __restrict__ bqkv, _Float16* __restrict__ Qh,
    _Float16* __restrict__ Kh, _Float16* __restrict__ Vt) {
  const int lane = threadIdx.x;
  const int nlo = lane & 15;
  const int hiK = lane >> 4;
  const int nt = blockIdx.x;  // 0..35  (N/64)
  const int mt = blockIdx.y;  // 0..127 (M/32)

  v8f acc[2][4];
#pragma unroll
  for (int r = 0; r < 2; ++r)
#pragma unroll
    for (int t = 0; t < 4; ++t) acc[r][t] = zero8f();

  const _Float16* arow0 = xh + (size_t)(mt * 32 + nlo) * Dd;
  const _Float16* arow1 = arow0 + (size_t)16 * Dd;
  const int nbase = nt * 64 + nlo;
  const _Float16* wbase = Wh + (size_t)nbase * Dd + hiK * 16;

  v16h Ab[2][2], Bbuf[2][4];
  auto load_chunk = [&](int k0, v16h A[2], v16h Bv[4]) {
    A[0] = a_op(arow0 + k0, hiK);
    A[1] = a_op(arow1 + k0, hiK);
#pragma unroll
    for (int t = 0; t < 4; ++t) Bv[t] = b_op(wbase + (size_t)(t * 16) * Dd + k0);
  };

  load_chunk(0, Ab[0], Bbuf[0]);
#pragma unroll
  for (int it = 0; it < KCH; ++it) {
    const int cur = it & 1;
    if (it + 1 < KCH) {
      __builtin_prefetch(wbase + (it + 2 < KCH ? (it + 2) * 32 : 0), 0, 3);
      load_chunk((it + 1) * 32, Ab[cur ^ 1], Bbuf[cur ^ 1]);
    }
    PIPELINE_FENCE();  // keep next-chunk loads issued before this chunk's WMMAs
#pragma unroll
    for (int t = 0; t < 4; ++t) {
      acc[0][t] = wmma16x16x32(Ab[cur][0], Bbuf[cur][t], acc[0][t]);
      acc[1][t] = wmma16x16x32(Ab[cur][1], Bbuf[cur][t], acc[1][t]);
    }
    PIPELINE_FENCE();  // keep this chunk's WMMAs before next iteration's loads
  }

#pragma unroll
  for (int r = 0; r < 2; ++r) {
#pragma unroll
    for (int t = 0; t < 4; ++t) {
      const int n = nbase + t * 16;
      const int h = n / 192, rem = n % 192;
      const int which = rem >> 6, dim = rem & 63;
      const float bias = bqkv[n];
#pragma unroll
      for (int j = 0; j < 8; ++j) {
        const int m = mt * 32 + r * 16 + hiK * 8 + j;  // global row b*S + s
        const int bI = m >> 11;
        const int s = m & (Ss - 1);
        const _Float16 hv = (_Float16)(acc[r][t][j] + bias);
        if (which == 2) {
          Vt[(((size_t)bI * Hh + h) * HD + dim) * Ss + s] = hv;
        } else {
          const size_t idx = (((size_t)bI * Hh + h) * Ss + s) * HD + dim;
          if (which == 0) Qh[idx] = hv; else Kh[idx] = hv;
        }
      }
    }
  }
}

// ============================================================================
// Kernel 2: banded attention. One wave per (b, h, 16-query tile).
// Phase 1: S = (Q K^T)/8 for 17 key tiles (2 WMMAs each, pipelined K loads).
// Phase 2: per-row max / exp / sum (v_exp_f32 + shfl_xor(16) reductions).
// Phase 3: O = P V, 9 K=32 WMMA chunks vs Vt, pipelined V loads; /= rowsum.
// ============================================================================
__global__ __launch_bounds__(32) void attn_kernel(
    const _Float16* __restrict__ Qh, const _Float16* __restrict__ Kh,
    const _Float16* __restrict__ Vt, const unsigned char* __restrict__ pad,
    _Float16* __restrict__ vals) {
  __shared__ float    sc[16][KEYS];      // 17 KB masked scores
  __shared__ _Float16 pbuf[16][288];     // 9 KB  exp(s - max), zero-padded
  __shared__ float    rowsum[16];

  const int lane = threadIdx.x;
  const int nlo = lane & 15;
  const int hiK = lane >> 4;
  const int qt = blockIdx.x;   // 0..127
  const int h  = blockIdx.y;   // 0..11
  const int bI = blockIdx.z;   // 0..1

  const _Float16* Qbh = Qh + ((size_t)bI * Hh + h) * Ss * HD;
  const _Float16* Kbh = Kh + ((size_t)bI * Hh + h) * Ss * HD;
  const _Float16* Vbh = Vt + ((size_t)bI * Hh + h) * HD * Ss;

  // Q A-operands (head dim 64 -> two K=32 chunks), lane row = query nlo
  const _Float16* qrow = Qbh + (size_t)(qt * 16 + nlo) * HD;
  const v16h a0 = a_op(qrow, hiK);
  const v16h a1 = a_op(qrow + 32, hiK);

  const int kbase = qt * 16 - WIN;
  const float NEG = -__builtin_inff();

  // ---- phase 1: scores (pipelined K-tile loads) ----
  v16h kb[2][2];
  auto load_k = [&](int kt, v16h dst[2]) {
    const int key = kbase + kt * 16 + nlo;
    if ((unsigned)key < (unsigned)Ss) {
      const _Float16* kp = Kbh + (size_t)key * HD + hiK * 16;
      dst[0] = b_op(kp);
      dst[1] = b_op(kp + 32);
    } else {
      dst[0] = zero16h();
      dst[1] = zero16h();
    }
  };
  load_k(0, kb[0]);
#pragma unroll
  for (int kt = 0; kt < KT; ++kt) {
    const int cur = kt & 1;
    if (kt + 1 < KT) load_k(kt + 1, kb[cur ^ 1]);
    PIPELINE_FENCE();
    v8f cc = zero8f();
    cc = wmma16x16x32(a0, kb[cur][0], cc);
    cc = wmma16x16x32(a1, kb[cur][1], cc);
    const int key = kbase + kt * 16 + nlo;
    const bool kvalid = ((unsigned)key < (unsigned)Ss) && (pad[bI * Ss + key] != 0);
#pragma unroll
    for (int j = 0; j < 8; ++j) {
      const int r = qt * 16 + hiK * 8 + j;
      int dd = r - key; dd = dd < 0 ? -dd : dd;
      const float s = (kvalid && dd <= WIN) ? cc[j] * 0.125f : NEG;
      sc[hiK * 8 + j][kt * 16 + nlo] = s;
    }
  }
  __syncthreads();

  // ---- phase 2: softmax numerators (row = nlo, halves split 272 keys) ----
  {
    const int row = nlo;
    const int off = hiK * 136;
    float mx = NEG;
    for (int i = 0; i < 136; ++i) mx = fmaxf(mx, sc[row][off + i]);
    mx = fmaxf(mx, __shfl_xor(mx, 16, 32));
    float sum = 0.0f;
    for (int i = 0; i < 136; ++i) {
      const float p = __expf(sc[row][off + i] - mx);
      sum += p;
      pbuf[row][off + i] = (_Float16)p;
    }
    sum += __shfl_xor(sum, 16, 32);
    if (hiK == 0) {
      rowsum[row] = sum;
#pragma unroll
      for (int i = 0; i < 16; ++i) pbuf[row][KEYS + i] = (_Float16)0.0f;  // pad
    }
  }
  __syncthreads();

  // ---- phase 3: O = P V (pipelined V-tile loads) ----
  v8f o[4] = {zero8f(), zero8f(), zero8f(), zero8f()};
  v16h vbuf[2][4];
  auto load_v = [&](int kc, v16h dst[4]) {
    const int kst = kbase + kc * 32 + hiK * 16;  // 16-aligned -> run all in/out
    const bool vinr = (unsigned)kst < (unsigned)Ss;
#pragma unroll
    for (int t = 0; t < 4; ++t)
      dst[t] = vinr ? b_op(Vbh + (size_t)(t * 16 + nlo) * Ss + kst) : zero16h();
  };
  load_v(0, vbuf[0]);
#pragma unroll
  for (int kc = 0; kc < 9; ++kc) {
    const int cur = kc & 1;
    if (kc + 1 < 9) load_v(kc + 1, vbuf[cur ^ 1]);
    PIPELINE_FENCE();
    // A-operand from P in LDS: two aligned 16-byte ds loads per lane
    const _Float16* prow = &pbuf[nlo][kc * 32 + hiK * 16];
    const v16h pa = cat(*(const v8h*)prow, *(const v8h*)(prow + 8));
#pragma unroll
    for (int t = 0; t < 4; ++t) o[t] = wmma16x16x32(pa, vbuf[cur][t], o[t]);
    PIPELINE_FENCE();
  }

  // epilogue: normalize, write vals[b*S+s][h*64+d] (A-matrix for out-proj)
#pragma unroll
  for (int t = 0; t < 4; ++t) {
#pragma unroll
    for (int j = 0; j < 8; ++j) {
      const int m = hiK * 8 + j;
      const float denom = rowsum[m];
      const float v = denom > 0.0f ? o[t][j] / denom : 0.0f;
      const size_t row = (size_t)bI * Ss + qt * 16 + m;
      vals[row * Dd + h * HD + t * 16 + nlo] = (_Float16)v;
    }
  }
}

// ============================================================================
// Kernel 3: out = vals @ Wo^T + bo   (M=4096, N=768, K=768), fp32 output.
// One wave per 32(M) x 64(N) tile; same 2-stage pipeline as kernel 1.
// ============================================================================
__global__ __launch_bounds__(32) void oproj_gemm_kernel(
    const _Float16* __restrict__ vals, const _Float16* __restrict__ Woh,
    const float* __restrict__ bo, float* __restrict__ out) {
  const int lane = threadIdx.x;
  const int nlo = lane & 15;
  const int hiK = lane >> 4;
  const int nt = blockIdx.x;  // 0..11  (N/64)
  const int mt = blockIdx.y;  // 0..127 (M/32)

  v8f acc[2][4];
#pragma unroll
  for (int r = 0; r < 2; ++r)
#pragma unroll
    for (int t = 0; t < 4; ++t) acc[r][t] = zero8f();

  const _Float16* arow0 = vals + (size_t)(mt * 32 + nlo) * Dd;
  const _Float16* arow1 = arow0 + (size_t)16 * Dd;
  const int nbase = nt * 64 + nlo;
  const _Float16* wbase = Woh + (size_t)nbase * Dd + hiK * 16;

  v16h Ab[2][2], Bbuf[2][4];
  auto load_chunk = [&](int k0, v16h A[2], v16h Bv[4]) {
    A[0] = a_op(arow0 + k0, hiK);
    A[1] = a_op(arow1 + k0, hiK);
#pragma unroll
    for (int t = 0; t < 4; ++t) Bv[t] = b_op(wbase + (size_t)(t * 16) * Dd + k0);
  };

  load_chunk(0, Ab[0], Bbuf[0]);
#pragma unroll
  for (int it = 0; it < KCH; ++it) {
    const int cur = it & 1;
    if (it + 1 < KCH) {
      __builtin_prefetch(wbase + (it + 2 < KCH ? (it + 2) * 32 : 0), 0, 3);
      load_chunk((it + 1) * 32, Ab[cur ^ 1], Bbuf[cur ^ 1]);
    }
    PIPELINE_FENCE();
#pragma unroll
    for (int t = 0; t < 4; ++t) {
      acc[0][t] = wmma16x16x32(Ab[cur][0], Bbuf[cur][t], acc[0][t]);
      acc[1][t] = wmma16x16x32(Ab[cur][1], Bbuf[cur][t], acc[1][t]);
    }
    PIPELINE_FENCE();
  }

#pragma unroll
  for (int r = 0; r < 2; ++r) {
#pragma unroll
    for (int t = 0; t < 4; ++t) {
      const int n = nbase + t * 16;
      const float bias = bo[n];
#pragma unroll
      for (int j = 0; j < 8; ++j) {
        const int m = mt * 32 + r * 16 + hiK * 8 + j;
        out[(size_t)m * Dd + n] = acc[r][t][j] + bias;
      }
    }
  }
}

extern "C" void kernel_launch(void* const* d_in, const int* in_sizes, int n_in,
                              void* d_out, int out_size, void* d_ws, size_t ws_size,
                              hipStream_t stream) {
  const float* x            = (const float*)d_in[0];
  const unsigned char* pad  = (const unsigned char*)d_in[1];  // bool mask
  const float* Wqkv         = (const float*)d_in[2];
  const float* bqkv         = (const float*)d_in[3];
  const float* Wo           = (const float*)d_in[4];
  const float* bo           = (const float*)d_in[5];
  float* out                = (float*)d_out;

  // workspace layout (all f16):
  //   Q, K:  [b,h,s,d]   Vt: [b,h,d,s]   vals: [b*s, e]
  //   xh:    [b*s, d]    Wqkvh: [3e, d]  Woh:  [e, e]
  const size_t nPer  = (size_t)Bb * Hh * Ss * HD;  // 3,145,728
  const size_t nX    = (size_t)Bb * Ss * Dd;       // 3,145,728
  const size_t nWqkv = (size_t)N3E * Dd;           // 1,769,472
  const size_t nWo   = (size_t)Dd * Dd;            //   589,824
  _Float16* Qh    = (_Float16*)d_ws;
  _Float16* Kh    = Qh + nPer;
  _Float16* Vt    = Kh + nPer;
  _Float16* vals  = Vt + nPer;
  _Float16* xh    = vals + nX;
  _Float16* Wqkvh = xh + nX;
  _Float16* Woh   = Wqkvh + nWqkv;
  (void)in_sizes; (void)n_in; (void)out_size; (void)ws_size;

  cvt_f16_kernel<<<(int)(nX / 8 / 256), 256, 0, stream>>>(x, xh, (int)nX);
  cvt_f16_kernel<<<(int)(nWqkv / 8 / 256), 256, 0, stream>>>(Wqkv, Wqkvh, (int)nWqkv);
  cvt_f16_kernel<<<(int)(nWo / 8 / 256), 256, 0, stream>>>(Wo, Woh, (int)nWo);

  qkv_gemm_kernel<<<dim3(N3E / 64, (Bb * Ss) / 32), 32, 0, stream>>>(
      xh, Wqkvh, bqkv, Qh, Kh, Vt);
  attn_kernel<<<dim3(Ss / 16, Hh, Bb), 32, 0, stream>>>(Qh, Kh, Vt, pad, vals);
  oproj_gemm_kernel<<<dim3(Dd / 64, (Bb * Ss) / 32), 32, 0, stream>>>(
      vals, Woh, bo, out);
}
